// DHSMoERBFDetector_3092376453875
// MI455X (gfx1250) — compile-verified, matrix operand order checked by reading, not run on hardware
//
#include <hip/hip_runtime.h>
#include <hip/hip_bf16.h>

// Problem constants (from reference)
#define BB 8192
#define DD 768
#define NC 4
#define RR 128
#define EE 20
#define HH 1024
#define KK (DD + NC * RR)   // 1280

typedef __attribute__((ext_vector_type(16))) __bf16        v16bf;
typedef __attribute__((ext_vector_type(8)))  float         v8f;
typedef __attribute__((ext_vector_type(4)))  unsigned int  v4u;

union Frag {
    v16bf v;
    v4u   q[2];
};

// ---------------------------------------------------------------------------
// Kernel 1: GaussianSmearing + concat(emb, rbf) -> X as bf16 [B, K]
// ---------------------------------------------------------------------------
__global__ __launch_bounds__(256) void build_x_kernel(
    const float* __restrict__ emb,        // [B, D]
    const float* __restrict__ feats,      // [NC, B]
    const float* __restrict__ rbf_off,    // [NC, R]
    const float* __restrict__ rbf_coeff,  // [NC]
    __hip_bfloat16* __restrict__ Xbf)     // [B, K]
{
    size_t i = (size_t)blockIdx.x * blockDim.x + threadIdx.x;
    if (i >= (size_t)BB * KK) return;
    int b = (int)(i / KK);
    int k = (int)(i % KK);
    float v;
    if (k < DD) {
        v = emb[(size_t)b * DD + k];
    } else {
        int kk = k - DD;
        int c  = kk >> 7;        // / 128
        int r  = kk & 127;       // % 128
        float d = feats[c * BB + b] - rbf_off[c * RR + r];
        v = __expf(rbf_coeff[c] * d * d);
    }
    Xbf[i] = __float2bfloat16(v);
}

// ---------------------------------------------------------------------------
// Kernel 2: W1[0] (K-major [K, H]) -> Wt bf16 [H, K] (read-coalesced)
// ---------------------------------------------------------------------------
__global__ __launch_bounds__(256) void transpose_w1_kernel(
    const float* __restrict__ W1,         // [E, K, H], expert 0
    __hip_bfloat16* __restrict__ Wt)      // [H, K]
{
    size_t i = (size_t)blockIdx.x * blockDim.x + threadIdx.x;
    if (i >= (size_t)KK * HH) return;
    int k = (int)(i / HH);
    int n = (int)(i % HH);
    Wt[(size_t)n * KK + k] = __float2bfloat16(W1[(size_t)k * HH + n]);
}

// ---------------------------------------------------------------------------
// Kernel 3: p[b] = b2[0]  (accumulator init, re-done every launch)
// ---------------------------------------------------------------------------
__global__ __launch_bounds__(256) void init_p_kernel(
    float* __restrict__ p, const float* __restrict__ b2)
{
    int b = blockIdx.x * blockDim.x + threadIdx.x;
    if (b < BB) p[b] = b2[0];
}

// ---------------------------------------------------------------------------
// Kernel 4: bf16 WMMA GEMM  h = relu(X @ W1_0 + b1_0), fused with the
//           second layer p[m] += sum_n h[m,n] * w2[n] (atomic fan-in).
// Grid: (H/256, B/64), block: 256 threads = 8 waves (2 M x 4 N per WG).
// Each wave: 32(M) x 64(N) tile = 8 accumulators (2 A frags x 4 B frags),
// K loop of 1280/32 = 40, software-pipelined (next-iter loads issued
// before current-iter WMMAs so s_wait_loadcnt overlaps 8 WMMAs of work).
// ---------------------------------------------------------------------------
__global__ __launch_bounds__(256) void gemm_fused_kernel(
    const __hip_bfloat16* __restrict__ Xbf,  // [B, K]
    const __hip_bfloat16* __restrict__ Wt,   // [H, K]
    const float* __restrict__ b1,            // [E, H] expert 0
    const float* __restrict__ W2,            // [E, H, 1] expert 0
    float* __restrict__ p)                   // [B]
{
    const int lane = threadIdx.x & 31;
    const int wave = threadIdx.x >> 5;
    const int mi = wave >> 2;                  // 0..1
    const int ni = wave & 3;                   // 0..3
    const int m0 = blockIdx.y * 64 + mi * 32;  // 32 M-rows per wave
    const int n0 = blockIdx.x * 256 + ni * 64; // 64 N-cols per wave

    const int half = lane >> 4;                // lane half selects K sub-block
    const int l16  = lane & 15;
    const int cSel = half * 8;                 // bf16 A/B layout: K = c..c+7, 16+c..

    v8f acc[2][4] = {};

    const __hip_bfloat16* aRow0 = Xbf + (size_t)(m0 + l16) * KK + cSel;
    const __hip_bfloat16* aRow1 = aRow0 + (size_t)16 * KK;
    const __hip_bfloat16* bBase = Wt  + (size_t)(n0 + l16) * KK + cSel;

    Frag a[2], b[4];
    // Prologue: fragments for k0 = 0
    a[0].q[0] = *(const v4u*)(aRow0);
    a[0].q[1] = *(const v4u*)(aRow0 + 16);
    a[1].q[0] = *(const v4u*)(aRow1);
    a[1].q[1] = *(const v4u*)(aRow1 + 16);
#pragma unroll
    for (int t = 0; t < 4; ++t) {
        const __hip_bfloat16* bRow = bBase + (size_t)t * 16 * KK;
        b[t].q[0] = *(const v4u*)(bRow);
        b[t].q[1] = *(const v4u*)(bRow + 16);
    }

    // Steady state: prefetch k0+32 fragments, then 8 WMMAs on k0 fragments.
    for (int k0 = 0; k0 < KK - 32; k0 += 32) {
        const int kn = k0 + 32;
        Frag an[2], bn[4];
        an[0].q[0] = *(const v4u*)(aRow0 + kn);
        an[0].q[1] = *(const v4u*)(aRow0 + kn + 16);
        an[1].q[0] = *(const v4u*)(aRow1 + kn);
        an[1].q[1] = *(const v4u*)(aRow1 + kn + 16);
#pragma unroll
        for (int t = 0; t < 4; ++t) {
            const __hip_bfloat16* bRow = bBase + (size_t)t * 16 * KK + kn;
            bn[t].q[0] = *(const v4u*)(bRow);
            bn[t].q[1] = *(const v4u*)(bRow + 16);
        }
        __builtin_prefetch(aRow0 + kn + 32, 0, 3);   // global_prefetch_b8 (near)
        __builtin_prefetch(aRow1 + kn + 32, 0, 3);

#pragma unroll
        for (int mt = 0; mt < 2; ++mt)
#pragma unroll
            for (int t = 0; t < 4; ++t)
                acc[mt][t] = __builtin_amdgcn_wmma_f32_16x16x32_bf16(
                    false, a[mt].v, false, b[t].v, (short)0, acc[mt][t],
                    false, false);
#pragma unroll
        for (int mt = 0; mt < 2; ++mt) a[mt] = an[mt];
#pragma unroll
        for (int t = 0; t < 4; ++t) b[t] = bn[t];
    }
    // Tail: last K step, no further loads.
#pragma unroll
    for (int mt = 0; mt < 2; ++mt)
#pragma unroll
        for (int t = 0; t < 4; ++t)
            acc[mt][t] = __builtin_amdgcn_wmma_f32_16x16x32_bf16(
                false, a[mt].v, false, b[t].v, (short)0, acc[mt][t],
                false, false);

    // Fused epilogue: bias + ReLU + dot with w2, reduce across N, atomic into p.
    float b1v[4], w2v[4];
#pragma unroll
    for (int t = 0; t < 4; ++t) {
        int n = n0 + t * 16 + l16;
        b1v[t] = b1[n];   // expert 0
        w2v[t] = W2[n];   // expert 0, [H,1]
    }
#pragma unroll
    for (int mt = 0; mt < 2; ++mt) {
#pragma unroll
        for (int r = 0; r < 8; ++r) {
            // C/D layout: VGPR r holds row M = r (lanes 0-15) / 8+r (lanes 16-31),
            // column N = lane & 15.
            float s = 0.f;
#pragma unroll
            for (int t = 0; t < 4; ++t) {
                float v = acc[mt][t][r] + b1v[t];
                v = v > 0.f ? v : 0.f;
                s += v * w2v[t];
            }
            s += __shfl_xor(s, 1, 32);
            s += __shfl_xor(s, 2, 32);
            s += __shfl_xor(s, 4, 32);
            s += __shfl_xor(s, 8, 32);
            if (l16 == 0) {
                atomicAdd(&p[m0 + mt * 16 + r + half * 8], s);
            }
        }
    }
}

// ---------------------------------------------------------------------------
// Kernel 5: stable-sort rank of component_idx + gather:
//   inv[b] = #{b' : cidx[b'] < cidx[b]} + #{b' < b : cidx[b'] == cidx[b]}
//   out[b] = p[inv[b]]
// Single 256-thread block, 32 contiguous elements per thread, LDS histograms.
// ---------------------------------------------------------------------------
__global__ __launch_bounds__(256) void rank_scatter_kernel(
    const int*   __restrict__ cidx,  // [B]
    const float* __restrict__ p,     // [B]
    float*       __restrict__ out)   // [B]
{
    __shared__ int cnt[256 * EE];
    __shared__ int classBase[EE + 1];
    const int t   = threadIdx.x;
    const int CH  = BB / 256;        // 32 elements per thread
    const int bse = t * CH;

    for (int c = 0; c < EE; ++c) cnt[t * EE + c] = 0;
    for (int i = 0; i < CH; ++i) cnt[t * EE + cidx[bse + i]]++;
    __syncthreads();

    if (t < EE) {   // exclusive scan across threads, per class
        int run = 0;
        for (int j = 0; j < 256; ++j) {
            int v = cnt[j * EE + t];
            cnt[j * EE + t] = run;
            run += v;
        }
        classBase[t + 1] = run;
    }
    __syncthreads();
    if (t == 0) {   // exclusive prefix over the 20 class totals
        classBase[0] = 0;
        for (int c = 0; c < EE; ++c) classBase[c + 1] += classBase[c];
    }
    __syncthreads();

    for (int i = 0; i < CH; ++i) {
        int e = bse + i;
        int c = cidx[e];
        int o = cnt[t * EE + c];
        cnt[t * EE + c] = o + 1;
        int rank = classBase[c] + o;   // == inv[e], in [0, B)
        out[e] = p[rank];
    }
}

// ---------------------------------------------------------------------------
extern "C" void kernel_launch(void* const* d_in, const int* in_sizes, int n_in,
                              void* d_out, int out_size, void* d_ws, size_t ws_size,
                              hipStream_t stream) {
    const float* emb       = (const float*)d_in[0];  // [B, D]
    const float* feats     = (const float*)d_in[1];  // [NC, B]
    const float* rbf_off   = (const float*)d_in[2];  // [NC, R]
    const float* rbf_coeff = (const float*)d_in[3];  // [NC]
    const float* W1        = (const float*)d_in[4];  // [E, K, H]
    const float* b1        = (const float*)d_in[5];  // [E, H]
    const float* W2        = (const float*)d_in[6];  // [E, H, 1]
    const float* b2        = (const float*)d_in[7];  // [E, 1]
    const int*   cidx      = (const int*)d_in[8];    // [B]
    float* out = (float*)d_out;

    // Workspace layout (bytes):
    //   Xbf : B*K bf16      = 20,971,520
    //   Wt  : H*K bf16      =  2,621,440
    //   p   : B   f32       =     32,768
    char* ws = (char*)d_ws;
    __hip_bfloat16* Xbf = (__hip_bfloat16*)ws;
    __hip_bfloat16* Wt  = (__hip_bfloat16*)(ws + (size_t)BB * KK * 2);
    float*          p   = (float*)(ws + (size_t)BB * KK * 2 + (size_t)HH * KK * 2);

    {
        size_t n = (size_t)BB * KK;
        build_x_kernel<<<(unsigned)((n + 255) / 256), 256, 0, stream>>>(
            emb, feats, rbf_off, rbf_coeff, Xbf);
    }
    {
        size_t n = (size_t)KK * HH;
        transpose_w1_kernel<<<(unsigned)((n + 255) / 256), 256, 0, stream>>>(W1, Wt);
    }
    init_p_kernel<<<BB / 256, 256, 0, stream>>>(p, b2);

    dim3 grid(HH / 256, BB / 64);   // (4, 128)
    gemm_fused_kernel<<<grid, 256, 0, stream>>>(Xbf, Wt, b1, W2, p);

    rank_scatter_kernel<<<1, 256, 0, stream>>>(cidx, p, out);
}
